// Global_MHRA_11020886081889
// MI455X (gfx1250) — matrix operand order repaired and checked by reading, not compiled, not run on hardware
//
#include <hip/hip_runtime.h>

typedef unsigned short u16;
typedef __attribute__((ext_vector_type(16))) __bf16 v16bf;
typedef __attribute__((ext_vector_type(8)))  float  v8f;
typedef __attribute__((ext_vector_type(4)))  int    v4i;

#define C_DIM   768
#define N_HEADS 12
#define HD      64
#define D_MLP_  3072
#define T_DIM   16
#define N_TOK   197
#define B_DIM   16
#define ROWS_Y  (B_DIM * T_DIM * N_TOK)   // 50432
#define KEYS    (T_DIM * N_TOK)           // 3152

#if __has_builtin(__builtin_amdgcn_global_load_async_to_lds_b128)
#define HAVE_ASYNC_LDS 1
#define AS1 __attribute__((address_space(1)))
#define AS3 __attribute__((address_space(3)))
#else
#define HAVE_ASYNC_LDS 0
#endif

__device__ __forceinline__ u16 f2bf(float f) {
    unsigned int u = __float_as_uint(f);
    u = (u + 0x7FFFu + ((u >> 16) & 1u)) >> 16;
    return (u16)u;
}
__device__ __forceinline__ float bf2f(u16 h) {
    return __uint_as_float(((unsigned int)h) << 16);
}

#if HAVE_ASYNC_LDS
__device__ __forceinline__ void wait_async0() {
#if __has_builtin(__builtin_amdgcn_s_wait_asynccnt)
    __builtin_amdgcn_s_wait_asynccnt(0);
#else
    asm volatile("s_wait_asynccnt 0" ::: "memory");
#endif
}
#endif

// ---------------- row-wise LayerNorm, one wave32 per row -------------------
template <bool BF16OUT>
__global__ __launch_bounds__(256) void ln_rows_kernel(
    const float* __restrict__ in, const float* __restrict__ w,
    const float* __restrict__ bias, float* __restrict__ out_f32,
    u16* __restrict__ out_bf16, int nrows) {
    int wave = threadIdx.x >> 5;
    int lane = threadIdx.x & 31;
    int row = blockIdx.x * 8 + wave;
    if (row >= nrows) return;
    const float* r = in + (size_t)row * C_DIM;
    float s = 0.f, ss = 0.f;
    for (int c = lane; c < C_DIM; c += 32) { float v = r[c]; s += v; ss += v * v; }
    for (int off = 16; off > 0; off >>= 1) {
        s  += __shfl_xor(s,  off, 32);
        ss += __shfl_xor(ss, off, 32);
    }
    float mu  = s * (1.f / C_DIM);
    float var = ss * (1.f / C_DIM) - mu * mu;
    float inv = rsqrtf(var + 1e-5f);
    for (int c = lane; c < C_DIM; c += 32) {
        float v = (r[c] - mu) * inv * w[c] + bias[c];
        if (BF16OUT) out_bf16[(size_t)row * C_DIM + c] = f2bf(v);
        else         out_f32 [(size_t)row * C_DIM + c] = v;
    }
}

// ---------------- convert Wk|Wv (rows 768..2303 of in_proj_w) to bf16 ------
__global__ void wkv_cvt_kernel(const float* __restrict__ w, u16* __restrict__ o) {
    int id = blockIdx.x * 256 + threadIdx.x;
    if (id < 2 * C_DIM * C_DIM) o[id] = f2bf(w[C_DIM * C_DIM + id]);
}

// ---------------- q = LN(x) @ Wq^T + bq (tiny: 16x768) ---------------------
__global__ __launch_bounds__(256) void qproj_kernel(
    const float* __restrict__ xn, const float* __restrict__ wq,
    const float* __restrict__ bq, float* __restrict__ q) {
    int id = blockIdx.x * 256 + threadIdx.x;           // 16*768
    int b = id / C_DIM, c = id % C_DIM;
    const float* xr = xn + (size_t)b * C_DIM;
    const float* wr = wq + (size_t)c * C_DIM;
    float s = bq[c];
    for (int k = 0; k < C_DIM; ++k) s += xr[k] * wr[k];
    q[id] = s;
}

// ---------------- depthwise 3x3x3 conv + residual, emits yk rows -----------
// yn layout: [bt=b*16+t][n][c] bf16 ; yk layout: [b*3152 + t*197 + n][c] bf16
__global__ __launch_bounds__(256) void dpe_conv_kernel(
    const u16* __restrict__ yn, const float* __restrict__ dw,
    const float* __restrict__ db, u16* __restrict__ yk) {
    int bid = blockIdx.x;                  // bt*197 + n
    int bt = bid / N_TOK, n = bid % N_TOK;
    int b = bt >> 4, t = bt & 15;
    size_t src = (size_t)bid * C_DIM;
    size_t dst = (size_t)(b * KEYS + t * N_TOK + n) * C_DIM;
    if (n == 0) {                          // cls token: passthrough
        for (int c = threadIdx.x; c < C_DIM; c += 256) yk[dst + c] = yn[src + c];
        return;
    }
    int p = n - 1, hh = p / 14, ww = p % 14;
    for (int c = threadIdx.x; c < C_DIM; c += 256) {
        float acc = db[c];
        const float* wc = dw + (size_t)c * 27;
        #pragma unroll
        for (int dt = 0; dt < 3; ++dt) {
            int tt = t + dt - 1; if (tt < 0 || tt > 15) continue;
            #pragma unroll
            for (int dh = 0; dh < 3; ++dh) {
                int ht = hh + dh - 1; if (ht < 0 || ht > 13) continue;
                #pragma unroll
                for (int dd = 0; dd < 3; ++dd) {
                    int wt = ww + dd - 1; if (wt < 0 || wt > 13) continue;
                    int n2 = 1 + ht * 14 + wt;
                    size_t si = (size_t)((b * 16 + tt) * N_TOK + n2) * C_DIM + c;
                    acc += bf2f(yn[si]) * wc[dt * 9 + dh * 3 + dd];
                }
            }
        }
        acc += bf2f(yn[src + c]);          // feats + f
        yk[dst + c] = f2bf(acc);
    }
}

// ---------------- K/V projection GEMM: [50432x768] x [768x1536] ------------
// WMMA bf16 -> f32, 128x128 block tile, 8 waves, wave owns 16x128 strip.
// Staging: GLOBAL_LOAD_ASYNC_TO_LDS_B128 (ASYNCcnt) with LDS double-buffer
// when available; else global_load_b128 + ds_store_b128.
__global__ __launch_bounds__(256) void kv_gemm_kernel(
    const u16* __restrict__ A,    // yk rows
    const u16* __restrict__ Bw,   // Wkv [1536][768] bf16
    const float* __restrict__ bias,
    u16* __restrict__ kout, u16* __restrict__ vout) {
#if HAVE_ASYNC_LDS
    __shared__ __align__(16) u16 As[2][128][40];   // 32 used + 8 pad
    __shared__ __align__(16) u16 Bs[2][128][40];
#else
    __shared__ __align__(16) u16 As[1][128][40];
    __shared__ __align__(16) u16 Bs[1][128][40];
#endif
    int tid = threadIdx.x;
    int wave = tid >> 5, lane = tid & 31;
    int m0 = blockIdx.x * 128;
    int n0 = blockIdx.y * 128;
    v8f acc[8];
    #pragma unroll
    for (int s = 0; s < 8; ++s)
        #pragma unroll
        for (int j = 0; j < 8; ++j) acc[s][j] = 0.f;

    int c0 = tid, c1 = tid + 256;          // two 16B chunks per thread/panel
    int ar0 = c0 >> 2, as0 = (c0 & 3) * 8;
    int ar1 = c1 >> 2, as1 = (c1 & 3) * 8;

#if HAVE_ASYNC_LDS
    // issue one panel's worth of async global->LDS copies into buffer `bsel`
    auto stage = [&](int kt, int bsel) {
        __builtin_amdgcn_global_load_async_to_lds_b128(
            (AS1 v4i*)(A + (size_t)(m0 + ar0) * C_DIM + kt + as0),
            (AS3 v4i*)&As[bsel][ar0][as0], 0, 0);
        __builtin_amdgcn_global_load_async_to_lds_b128(
            (AS1 v4i*)(A + (size_t)(m0 + ar1) * C_DIM + kt + as1),
            (AS3 v4i*)&As[bsel][ar1][as1], 0, 0);
        __builtin_amdgcn_global_load_async_to_lds_b128(
            (AS1 v4i*)(Bw + (size_t)(n0 + ar0) * C_DIM + kt + as0),
            (AS3 v4i*)&Bs[bsel][ar0][as0], 0, 0);
        __builtin_amdgcn_global_load_async_to_lds_b128(
            (AS1 v4i*)(Bw + (size_t)(n0 + ar1) * C_DIM + kt + as1),
            (AS3 v4i*)&Bs[bsel][ar1][as1], 0, 0);
    };
    stage(0, 0);
    for (int kt = 0, it = 0; kt < C_DIM; kt += 32, ++it) {
        int cur = it & 1;
        wait_async0();          // own panel copies complete
        __syncthreads();        // everyone's copies visible; prev compute done
        if (kt + 32 < C_DIM) stage(kt + 32, cur ^ 1);

        union { v16bf v; uint4 q[2]; } af;
        int am = wave * 16 + (lane & 15);
        int ak = (lane >> 4) * 8;
        af.q[0] = *(const uint4*)&As[cur][am][ak];
        af.q[1] = *(const uint4*)&As[cur][am][ak + 16];
        #pragma unroll
        for (int s = 0; s < 8; ++s) {
            union { v16bf v; uint4 q[2]; } bf_;
            int bn = s * 16 + (lane & 15);
            int bk = (lane >> 4) * 8;
            bf_.q[0] = *(const uint4*)&Bs[cur][bn][bk];
            bf_.q[1] = *(const uint4*)&Bs[cur][bn][bk + 16];
            acc[s] = __builtin_amdgcn_wmma_f32_16x16x32_bf16(
                false, af.v, false, bf_.v, (short)0, acc[s], false, false);
        }
    }
#else
    for (int kt = 0; kt < C_DIM; kt += 32) {
        uint4 ra0 = *(const uint4*)(A  + (size_t)(m0 + ar0) * C_DIM + kt + as0);
        uint4 ra1 = *(const uint4*)(A  + (size_t)(m0 + ar1) * C_DIM + kt + as1);
        uint4 rb0 = *(const uint4*)(Bw + (size_t)(n0 + ar0) * C_DIM + kt + as0);
        uint4 rb1 = *(const uint4*)(Bw + (size_t)(n0 + ar1) * C_DIM + kt + as1);
        __syncthreads();
        *(uint4*)&As[0][ar0][as0] = ra0;
        *(uint4*)&As[0][ar1][as1] = ra1;
        *(uint4*)&Bs[0][ar0][as0] = rb0;
        *(uint4*)&Bs[0][ar1][as1] = rb1;
        __syncthreads();

        union { v16bf v; uint4 q[2]; } af;
        int am = wave * 16 + (lane & 15);
        int ak = (lane >> 4) * 8;
        af.q[0] = *(const uint4*)&As[0][am][ak];
        af.q[1] = *(const uint4*)&As[0][am][ak + 16];
        #pragma unroll
        for (int s = 0; s < 8; ++s) {
            union { v16bf v; uint4 q[2]; } bf_;
            int bn = s * 16 + (lane & 15);
            int bk = (lane >> 4) * 8;
            bf_.q[0] = *(const uint4*)&Bs[0][bn][bk];
            bf_.q[1] = *(const uint4*)&Bs[0][bn][bk + 16];
            acc[s] = __builtin_amdgcn_wmma_f32_16x16x32_bf16(
                false, af.v, false, bf_.v, (short)0, acc[s], false, false);
        }
    }
#endif

    // epilogue: D[v][lane] = (M = v + 8*(lane>>4), N = lane&15)
    #pragma unroll
    for (int s = 0; s < 8; ++s) {
        int ng = n0 + s * 16 + (lane & 15);
        float bs = bias[C_DIM + ng];
        #pragma unroll
        for (int i = 0; i < 8; ++i) {
            int mg = m0 + wave * 16 + i + (lane >> 4) * 8;
            u16 o = f2bf(acc[s][i] + bs);
            size_t off = (size_t)mg * C_DIM;
            if (ng < C_DIM) kout[off + ng] = o;
            else            vout[off + ng - C_DIM] = o;
        }
    }
}

// ---------------- attention: one block per (b,h), 3152 keys ----------------
__global__ __launch_bounds__(256) void attn_kernel(
    const float* __restrict__ q, const u16* __restrict__ kb,
    const u16* __restrict__ vb, float* __restrict__ outp) {
    int b = blockIdx.x / N_HEADS, h = blockIdx.x % N_HEADS;
    __shared__ float logits[KEYS];
    __shared__ float qv[HD];
    __shared__ float red[256];
    int tid = threadIdx.x;
    if (tid < HD) qv[tid] = q[b * C_DIM + h * HD + tid];
    __syncthreads();
    for (int j = tid; j < KEYS; j += 256) {
        const u16* kr = kb + (size_t)(b * KEYS + j) * C_DIM + h * HD;
        if (j + 256 < KEYS)
            __builtin_prefetch((const void*)(kr + (size_t)256 * C_DIM), 0, 1);
        float s = 0.f;
        #pragma unroll
        for (int d = 0; d < HD; ++d) s += qv[d] * bf2f(kr[d]);
        logits[j] = s * 0.125f;            // 1/sqrt(64)
    }
    __syncthreads();
    float m = -1e30f;
    for (int j = tid; j < KEYS; j += 256) m = fmaxf(m, logits[j]);
    red[tid] = m; __syncthreads();
    for (int s = 128; s > 0; s >>= 1) {
        if (tid < s) red[tid] = fmaxf(red[tid], red[tid + s]);
        __syncthreads();
    }
    float M = red[0]; __syncthreads();
    float ssum = 0.f;
    for (int j = tid; j < KEYS; j += 256) {
        float e = __expf(logits[j] - M); logits[j] = e; ssum += e;
    }
    red[tid] = ssum; __syncthreads();
    for (int s = 128; s > 0; s >>= 1) {
        if (tid < s) red[tid] += red[tid + s];
        __syncthreads();
    }
    float inv = 1.f / red[0]; __syncthreads();
    int d = tid & 63, g = tid >> 6;        // 4 groups over keys
    float acc = 0.f;
    for (int j = g; j < KEYS; j += 4)
        acc += logits[j] * bf2f(vb[(size_t)(b * KEYS + j) * C_DIM + h * HD + d]);
    red[tid] = acc; __syncthreads();
    if (g == 0)
        outp[b * C_DIM + h * HD + d] =
            (red[d] + red[64 + d] + red[128 + d] + red[192 + d]) * inv;
}

// ---------------- out-proj + residual --------------------------------------
__global__ __launch_bounds__(256) void outproj_kernel(
    const float* __restrict__ attn, const float* __restrict__ ow,
    const float* __restrict__ ob, const float* __restrict__ x,
    float* __restrict__ x1) {
    int id = blockIdx.x * 256 + threadIdx.x;
    int b = id / C_DIM, c = id % C_DIM;
    const float* ar = attn + (size_t)b * C_DIM;
    const float* wr = ow + (size_t)c * C_DIM;
    float s = ob[c];
    for (int k = 0; k < C_DIM; ++k) s += ar[k] * wr[k];
    x1[id] = x[id] + s;
}

// ---------------- MLP fc + exact gelu --------------------------------------
__global__ __launch_bounds__(256) void fc_gelu_kernel(
    const float* __restrict__ h, const float* __restrict__ fw,
    const float* __restrict__ fb, float* __restrict__ g) {
    int id = blockIdx.x * 256 + threadIdx.x;   // 16*3072
    int b = id / D_MLP_, m = id % D_MLP_;
    const float* hr = h + (size_t)b * C_DIM;
    const float* wr = fw + (size_t)m * C_DIM;
    float s = fb[m];
    for (int k = 0; k < C_DIM; ++k) s += hr[k] * wr[k];
    g[id] = 0.5f * s * (1.f + erff(s * 0.70710678118654752f));
}

// ---------------- MLP proj + final residual --------------------------------
__global__ __launch_bounds__(256) void proj_final_kernel(
    const float* __restrict__ g, const float* __restrict__ pw,
    const float* __restrict__ pb, const float* __restrict__ x1,
    float* __restrict__ out) {
    int id = blockIdx.x * 256 + threadIdx.x;   // 16*768
    int b = id / C_DIM, c = id % C_DIM;
    const float* gr = g + (size_t)b * D_MLP_;
    const float* wr = pw + (size_t)c * D_MLP_;
    float s = pb[c];
    for (int k = 0; k < D_MLP_; ++k) s += gr[k] * wr[k];
    out[id] = x1[id] + s;
}

extern "C" void kernel_launch(void* const* d_in, const int* in_sizes, int n_in,
                              void* d_out, int out_size, void* d_ws, size_t ws_size,
                              hipStream_t stream) {
    (void)in_sizes; (void)n_in; (void)out_size; (void)ws_size;
    const float* x         = (const float*)d_in[0];
    const float* y         = (const float*)d_in[1];
    /* d_in[2] = T (always 16) */
    const float* dpe_w     = (const float*)d_in[3];
    const float* dpe_b     = (const float*)d_in[4];
    const float* in_proj_w = (const float*)d_in[5];
    const float* in_proj_b = (const float*)d_in[6];
    const float* out_w     = (const float*)d_in[7];
    const float* out_b     = (const float*)d_in[8];
    const float* ln1_w     = (const float*)d_in[9];
    const float* ln1_b     = (const float*)d_in[10];
    const float* ln2_w     = (const float*)d_in[11];
    const float* ln2_b     = (const float*)d_in[12];
    const float* ln3_w     = (const float*)d_in[13];
    const float* ln3_b     = (const float*)d_in[14];
    const float* fc_w      = (const float*)d_in[15];
    const float* fc_b      = (const float*)d_in[16];
    const float* proj_w    = (const float*)d_in[17];
    const float* proj_b    = (const float*)d_in[18];
    float* out = (float*)d_out;

    char* ws = (char*)d_ws;
    size_t off = 0;
    auto take = [&](size_t bytes) {
        char* p = ws + off;
        off += (bytes + 255) & ~(size_t)255;
        return p;
    };
    u16*  yn   = (u16*)take((size_t)ROWS_Y * C_DIM * 2);
    u16*  yk   = (u16*)take((size_t)ROWS_Y * C_DIM * 2);
    u16*  kb   = (u16*)take((size_t)ROWS_Y * C_DIM * 2);
    u16*  vb   = (u16*)take((size_t)ROWS_Y * C_DIM * 2);
    u16*  wkv  = (u16*)take((size_t)2 * C_DIM * C_DIM * 2);
    float* qb   = (float*)take(B_DIM * C_DIM * 4);
    float* xn   = (float*)take(B_DIM * C_DIM * 4);
    float* attn = (float*)take(B_DIM * C_DIM * 4);
    float* x1   = (float*)take(B_DIM * C_DIM * 4);
    float* hbuf = (float*)take(B_DIM * C_DIM * 4);
    float* gbuf = (float*)take(B_DIM * D_MLP_ * 4);

    ln_rows_kernel<true><<<(ROWS_Y + 7) / 8, 256, 0, stream>>>(
        y, ln3_w, ln3_b, (float*)nullptr, yn, ROWS_Y);
    ln_rows_kernel<false><<<2, 256, 0, stream>>>(
        x, ln1_w, ln1_b, xn, (u16*)nullptr, B_DIM);
    wkv_cvt_kernel<<<(2 * C_DIM * C_DIM + 255) / 256, 256, 0, stream>>>(in_proj_w, wkv);
    qproj_kernel<<<(B_DIM * C_DIM) / 256, 256, 0, stream>>>(xn, in_proj_w, in_proj_b, qb);
    dpe_conv_kernel<<<ROWS_Y, 256, 0, stream>>>(yn, dpe_w, dpe_b, yk);
    kv_gemm_kernel<<<dim3(ROWS_Y / 128, (2 * C_DIM) / 128), 256, 0, stream>>>(
        yk, wkv, in_proj_b, kb, vb);
    attn_kernel<<<B_DIM * N_HEADS, 256, 0, stream>>>(qb, kb, vb, attn);
    outproj_kernel<<<(B_DIM * C_DIM) / 256, 256, 0, stream>>>(attn, out_w, out_b, x, x1);
    ln_rows_kernel<false><<<2, 256, 0, stream>>>(
        x1, ln2_w, ln2_b, hbuf, (u16*)nullptr, B_DIM);
    fc_gelu_kernel<<<(B_DIM * D_MLP_) / 256, 256, 0, stream>>>(hbuf, fc_w, fc_b, gbuf);
    proj_final_kernel<<<(B_DIM * C_DIM) / 256, 256, 0, stream>>>(gbuf, proj_w, proj_b, x1, out);
}
